// _Fact_tt_Attention_21191368638970
// MI455X (gfx1250) — compile-verified, matrix operand order checked by reading, not run on hardware
//
#include <hip/hip_runtime.h>
#include <stdint.h>

// ---------------------------------------------------------------------------
// CDNA5 (gfx1250) implementation of FacT-tuned windowed attention.
// wave32, WMMA 16x16x32 bf16 with fp32 accumulation.
//  - GEMMs: 16x64 strip per wave, double-buffered fragment pipeline
//  - Attention: flash-style, 32 q-rows per wave (K/V fragments reused 2x)
// ---------------------------------------------------------------------------

typedef __attribute__((ext_vector_type(16))) __bf16 v16bf;
typedef __attribute__((ext_vector_type(8)))  float  v8f;

#define NH    12
#define HD    64
#define HG    48
#define NTOK  2304        // 48*48
#define CH    768
#define RANKL 16
#define FACT_SCALE 1.0f
#define NQT2  (NTOK/32)   // 72 q tiles of 32 rows

__device__ __forceinline__ unsigned short f2bf(float f) {
  unsigned u = __float_as_uint(f);
  u += 0x7FFFu + ((u >> 16) & 1u);   // round-to-nearest-even
  return (unsigned short)(u >> 16);
}

union Frag {            // 16 bf16 elements = 8 VGPRs, matches WMMA A/B operand
  v16bf   v;
  uint32_t w[8];
  uint4    q[2];
};

__device__ __forceinline__ v8f wmma_bf16(const Frag& a, const Frag& b, v8f c) {
  return __builtin_amdgcn_wmma_f32_16x16x32_bf16(
      /*neg_a=*/false, a.v, /*neg_b=*/false, b.v,
      /*c_mod=*/(short)0, c, /*reuse_a=*/false, /*reuse_b=*/false);
}

// ---------------------------------------------------------------------------
// fp32 -> bf16 elementwise
__global__ void k_cvt_bf16(const float* __restrict__ src,
                           unsigned short* __restrict__ dst, int n) {
  int i = blockIdx.x * 256 + threadIdx.x;
  if (i < n) dst[i] = f2bf(src[i]);
}

// ---------------------------------------------------------------------------
// Pack a row-major fp32 B matrix (K x N) into per-wave WMMA B-fragment layout.
// Word index = ((kt*(N/16)+nt)*32 + lane)*8 + j.
// lane&15 -> column within tile; lane>>4 selects K sub-block (ISA layout):
//   j<4 : k = kb + 2j (+1 in high half of the dword)
//   j>=4: k = 16 + kb + 2(j-4)
__global__ void k_packB(const float* __restrict__ B, uint32_t* __restrict__ P,
                        int K, int N) {
  int idx = blockIdx.x * 256 + threadIdx.x;
  int total = (K / 32) * (N / 16) * 256;
  if (idx >= total) return;
  int j    = idx & 7;
  int lane = (idx >> 3) & 31;
  int rest = idx >> 8;
  int ntiles = N >> 4;
  int nt = rest % ntiles;
  int kt = rest / ntiles;
  int n  = nt * 16 + (lane & 15);
  int kb = (lane >> 4) * 8;
  int kk = kt * 32 + ((j < 4) ? (kb + 2 * j) : (16 + kb + 2 * (j - 4)));
  uint32_t lo = f2bf(B[(size_t)kk * N + n]);
  uint32_t hi = f2bf(B[(size_t)(kk + 1) * N + n]);
  P[idx] = lo | (hi << 16);
}

// ---------------------------------------------------------------------------
// C(MxN,f32) = A(MxK, bf16 row-major) * Bpacked + bias[N].
// One wave computes a 16x64 strip (4 tiles). Fragment sets are double-buffered
// so the loads for K-step k+1 are in flight during the 4 WMMAs of step k.
__global__ void __launch_bounds__(128)
k_gemm4(const unsigned short* __restrict__ A, const uint32_t* __restrict__ Bp,
        const float* __restrict__ bias, float* __restrict__ Cout,
        int M, int N, int K) {
  int lane = threadIdx.x & 31;
  int wave = (blockIdx.x * 128 + threadIdx.x) >> 5;
  int ngroups = N >> 6;           // groups of 4 adjacent 16-wide tiles
  int mt = wave / ngroups;
  int ng = wave % ngroups;
  int ntiles = N >> 4;
  int r  = lane & 15;
  int hi = lane >> 4;
  int kb = hi * 8;

  v8f acc[4];
  #pragma unroll
  for (int t = 0; t < 4; ++t) {
    float bv = bias[(ng * 4 + t) * 16 + r];
    #pragma unroll
    for (int j = 0; j < 8; ++j) acc[t][j] = bv;
  }

  const unsigned short* Arow = A + (size_t)(mt * 16 + r) * K;
  const uint32_t* Bbase = Bp + (size_t)(ng * 4) * 256 + (size_t)lane * 8;
  size_t bstride = (size_t)ntiles * 256;
  const int nk = K / 32;          // even for all our shapes (24)

  Frag a[2];
  Frag b[2][4];
  auto loadset = [&](int buf, int kt2) {
    const unsigned short* Ak = Arow + kt2 * 32;
    a[buf].q[0] = *(const uint4*)(Ak + kb);
    a[buf].q[1] = *(const uint4*)(Ak + 16 + kb);
    const uint32_t* Bk = Bbase + (size_t)kt2 * bstride;
    __builtin_prefetch(Bk + 2 * bstride, 0, 1);
    #pragma unroll
    for (int t = 0; t < 4; ++t) {
      b[buf][t].q[0] = *(const uint4*)(Bk + t * 256);
      b[buf][t].q[1] = *(const uint4*)(Bk + t * 256 + 4);
    }
  };
  auto compute = [&](int buf) {
    #pragma unroll
    for (int t = 0; t < 4; ++t) acc[t] = wmma_bf16(a[buf], b[buf][t], acc[t]);
  };

  loadset(0, 0);
  int kt = 0;
  for (; kt + 2 < nk; kt += 2) {
    loadset(1, kt + 1);   // in flight during compute(0)
    compute(0);
    loadset(0, kt + 2);   // in flight during compute(1)
    compute(1);
  }
  loadset(1, kt + 1);
  compute(0);
  compute(1);

  // C layout: row m = j + 8*hi, col = r (within each 16-wide tile)
  float* Crow = Cout + (size_t)(mt * 16) * N + (size_t)ng * 64;
  #pragma unroll
  for (int t = 0; t < 4; ++t)
    #pragma unroll
    for (int j = 0; j < 8; ++j)
      Crow[(size_t)(j + 8 * hi) * N + t * 16 + r] = acc[t][j];
}

// ---------------------------------------------------------------------------
// u = x @ FacTu  (2304x768 @ 768x16)
__global__ void k_u(const float* __restrict__ x, const float* __restrict__ Fu,
                    float* __restrict__ u) {
  int idx = blockIdx.x * 256 + threadIdx.x;
  if (idx >= NTOK * RANKL) return;
  int t = idx / RANKL, rr = idx % RANKL;
  const float* xr = x + (size_t)t * CH;
  float s = 0.f;
  for (int i = 0; i < CH; ++i) s += xr[i] * Fu[i * RANKL + rr];
  u[idx] = s;
}

// uq = u @ q_facts ; uv = u @ v_facts  (16x16 each)
__global__ void k_uf(const float* __restrict__ u, const float* __restrict__ qf,
                     const float* __restrict__ vf, float* __restrict__ uq,
                     float* __restrict__ uv) {
  int idx = blockIdx.x * 256 + threadIdx.x;
  if (idx >= NTOK * RANKL) return;
  int t = idx / RANKL, rr = idx % RANKL;
  const float* ur = u + (size_t)t * RANKL;
  float sq = 0.f, sv = 0.f;
  for (int i = 0; i < RANKL; ++i) {
    sq += ur[i] * qf[i * RANKL + rr];
    sv += ur[i] * vf[i * RANKL + rr];
  }
  uq[idx] = sq;
  uv[idx] = sv;
}

// qkv[:, :C] += s*(uq @ FacTv) ; qkv[:, 2C:] += s*(uv @ FacTv)
__global__ void k_fact_add(const float* __restrict__ uq,
                           const float* __restrict__ uv,
                           const float* __restrict__ Fv,
                           float* __restrict__ qkv) {
  int idx = blockIdx.x * 256 + threadIdx.x;
  if (idx >= NTOK * CH) return;
  int t = idx / CH, c = idx % CH;
  const float* a = uq + (size_t)t * RANKL;
  const float* b = uv + (size_t)t * RANKL;
  float sq = 0.f, sv = 0.f;
  for (int r = 0; r < RANKL; ++r) {
    float f = Fv[r * CH + c];
    sq += a[r] * f;
    sv += b[r] * f;
  }
  qkv[(size_t)t * (3 * CH) + c]          += FACT_SCALE * sq;
  qkv[(size_t)t * (3 * CH) + 2 * CH + c] += FACT_SCALE * sv;
}

// Split qkv into head-major bf16 tensors: Qh,Kh (12,2304,64) and Vt (12,64,2304)
__global__ void k_pack_qkv(const float* __restrict__ qkv,
                           unsigned short* __restrict__ Qh,
                           unsigned short* __restrict__ Kh,
                           unsigned short* __restrict__ Vt) {
  int idx = blockIdx.x * 256 + threadIdx.x;
  if (idx >= NH * NTOK * HD) return;
  int c = idx % HD;
  int t = (idx / HD) % NTOK;
  int h = idx / (HD * NTOK);
  size_t base = (size_t)t * (3 * CH) + h * HD + c;
  Qh[((size_t)h * NTOK + t) * HD + c] = f2bf(qkv[base]);
  Kh[((size_t)h * NTOK + t) * HD + c] = f2bf(qkv[base + CH]);
  Vt[((size_t)h * HD + c) * NTOK + t] = f2bf(qkv[base + 2 * CH]);
}

// Decomposed rel-pos bias tables:
//   relh[h,qh,qw,kh] = sum_c q[h,t,c] * rel_pos_h[qh-kh+47, c]
//   relw[h,qh,qw,kw] = sum_c q[h,t,c] * rel_pos_w[qw-kw+47, c]
__global__ void k_rel(const float* __restrict__ qkv,
                      const float* __restrict__ rph,
                      const float* __restrict__ rpw,
                      float* __restrict__ relh, float* __restrict__ relw) {
  int idx = blockIdx.x * 256 + threadIdx.x;
  if (idx >= NH * HG * HG * HG) return;
  int k  = idx % HG;
  int qw = (idx / HG) % HG;
  int qh = (idx / (HG * HG)) % HG;
  int h  = idx / (HG * HG * HG);
  int t  = qh * HG + qw;
  const float* q  = qkv + (size_t)t * (3 * CH) + h * HD;  // q slice, post-FacT
  const float* ph = rph + (size_t)(qh - k + HG - 1) * HD;
  const float* pw = rpw + (size_t)(qw - k + HG - 1) * HD;
  float sh = 0.f, sw = 0.f;
  for (int c = 0; c < HD; ++c) {
    float qv = q[c];
    sh += qv * ph[c];
    sw += qv * pw[c];
  }
  relh[idx] = sh;
  relw[idx] = sw;
}

// ---------------------------------------------------------------------------
// Fused flash-style attention. One wave per (head, 32-row q tile).
// Per 32-key step: 8 WMMAs for S (two 16-row subtiles sharing K fragments),
// online softmax, P routed C->A layout via LDS (32x32 bf16 per wave),
// 8 WMMAs for O += P @ V (V fragments shared by both subtiles).
__global__ void __launch_bounds__(128)
k_attn(const unsigned short* __restrict__ Qh, const unsigned short* __restrict__ Kh,
       const unsigned short* __restrict__ Vt, const float* __restrict__ relh,
       const float* __restrict__ relw, float* __restrict__ out) {
  __shared__ __align__(16) unsigned short lds[4][32 * 32];
  const int lane = threadIdx.x & 31;
  const int wib  = threadIdx.x >> 5;
  const int wave = blockIdx.x * 4 + wib;
  const int h  = wave / NQT2;
  const int qt = wave % NQT2;       // 32-row tile index
  const int r  = lane & 15;
  const int hi = lane >> 4;
  const int kb = hi * 8;
  const float scale = 0.125f;       // 64^-0.5

  // Persistent Q A-fragments: [subtile][channel chunk 0..31 / 32..63]
  const unsigned short* Qb = Qh + ((size_t)h * NTOK + (size_t)qt * 32) * HD;
  Frag aq[2][2];
  #pragma unroll
  for (int qs = 0; qs < 2; ++qs) {
    const unsigned short* qr = Qb + (size_t)(qs * 16 + r) * HD;
    aq[qs][0].q[0] = *(const uint4*)(qr + kb);
    aq[qs][0].q[1] = *(const uint4*)(qr + 16 + kb);
    aq[qs][1].q[0] = *(const uint4*)(qr + 32 + kb);
    aq[qs][1].q[1] = *(const uint4*)(qr + 48 + kb);
  }

  int rowbase[2][8];
  float mrow[2][8], lrow[2][8];
  v8f o[2][4];
  #pragma unroll
  for (int qs = 0; qs < 2; ++qs)
    #pragma unroll
    for (int j = 0; j < 8; ++j) {
      rowbase[qs][j] = (h * NTOK + qt * 32 + qs * 16 + j + 8 * hi) * HG;
      mrow[qs][j] = -1e30f;
      lrow[qs][j] = 0.f;
      o[qs][0][j] = 0.f; o[qs][1][j] = 0.f; o[qs][2][j] = 0.f; o[qs][3][j] = 0.f;
    }

  const unsigned short* Kb = Kh + (size_t)h * NTOK * HD;
  const unsigned short* Vb = Vt + (size_t)h * HD * NTOK;
  unsigned short* myl = lds[wib];
  const uint32_t* mylw = (const uint32_t*)myl;

  for (int kt = 0; kt < NTOK / 32; ++kt) {
    const int key0 = kt * 32 + r;    // this lane's key column, tile 0
    const int key1 = key0 + 16;      // tile 1

    // K fragments: [key tile][channel chunk] -- shared by both q subtiles
    Frag bk[2][2];
    const unsigned short* k0p = Kb + (size_t)key0 * HD;
    const unsigned short* k1p = Kb + (size_t)key1 * HD;
    bk[0][0].q[0] = *(const uint4*)(k0p + kb);
    bk[0][0].q[1] = *(const uint4*)(k0p + 16 + kb);
    bk[0][1].q[0] = *(const uint4*)(k0p + 32 + kb);
    bk[0][1].q[1] = *(const uint4*)(k0p + 48 + kb);
    bk[1][0].q[0] = *(const uint4*)(k1p + kb);
    bk[1][0].q[1] = *(const uint4*)(k1p + 16 + kb);
    bk[1][1].q[0] = *(const uint4*)(k1p + 32 + kb);
    bk[1][1].q[1] = *(const uint4*)(k1p + 48 + kb);

    v8f s[2][2];
    #pragma unroll
    for (int qs = 0; qs < 2; ++qs)
      #pragma unroll
      for (int ktile = 0; ktile < 2; ++ktile) {
        v8f z;
        #pragma unroll
        for (int j = 0; j < 8; ++j) z[j] = 0.f;
        z = wmma_bf16(aq[qs][0], bk[ktile][0], z);
        z = wmma_bf16(aq[qs][1], bk[ktile][1], z);
        s[qs][ktile] = z;
      }

    // scale + decomposed rel-pos bias, then online softmax per subtile
    const int kh0 = key0 / HG, kw0 = key0 - kh0 * HG;
    const int kh1 = key1 / HG, kw1 = key1 - kh1 * HG;
    #pragma unroll
    for (int qs = 0; qs < 2; ++qs) {
      float mx[8];
      #pragma unroll
      for (int j = 0; j < 8; ++j) {
        float b0 = relh[rowbase[qs][j] + kh0] + relw[rowbase[qs][j] + kw0];
        float b1 = relh[rowbase[qs][j] + kh1] + relw[rowbase[qs][j] + kw1];
        s[qs][0][j] = s[qs][0][j] * scale + b0;
        s[qs][1][j] = s[qs][1][j] * scale + b1;
        mx[j] = fmaxf(s[qs][0][j], s[qs][1][j]);
      }
      #pragma unroll
      for (int j = 0; j < 8; ++j) {
        float v = mx[j];
        v = fmaxf(v, __shfl_xor(v, 1, 32));
        v = fmaxf(v, __shfl_xor(v, 2, 32));
        v = fmaxf(v, __shfl_xor(v, 4, 32));
        v = fmaxf(v, __shfl_xor(v, 8, 32));
        mx[j] = v;
      }
      #pragma unroll
      for (int j = 0; j < 8; ++j) {
        float mn = fmaxf(mrow[qs][j], mx[j]);
        float alpha = __expf(mrow[qs][j] - mn);
        mrow[qs][j] = mn;
        s[qs][0][j] = __expf(s[qs][0][j] - mn);
        s[qs][1][j] = __expf(s[qs][1][j] - mn);
        float sum = s[qs][0][j] + s[qs][1][j];
        sum += __shfl_xor(sum, 1, 32);
        sum += __shfl_xor(sum, 2, 32);
        sum += __shfl_xor(sum, 4, 32);
        sum += __shfl_xor(sum, 8, 32);
        lrow[qs][j] = lrow[qs][j] * alpha + sum;
        o[qs][0][j] *= alpha; o[qs][1][j] *= alpha;
        o[qs][2][j] *= alpha; o[qs][3][j] *= alpha;
      }
      // P (C layout) -> LDS row-major 32x32 bf16
      #pragma unroll
      for (int j = 0; j < 8; ++j) {
        myl[(qs * 16 + j + 8 * hi) * 32 + r]      = f2bf(s[qs][0][j]);
        myl[(qs * 16 + j + 8 * hi) * 32 + 16 + r] = f2bf(s[qs][1][j]);
      }
    }
    asm volatile("s_wait_dscnt 0" ::: "memory");

    // Re-read P in A-fragment layout, one fragment per subtile
    Frag ap[2];
    #pragma unroll
    for (int qs = 0; qs < 2; ++qs)
      #pragma unroll
      for (int j = 0; j < 8; ++j) {
        int row = qs * 16 + r;
        int widx = (j < 4) ? (row * 16 + (kb >> 1) + j)
                           : (row * 16 + 8 + (kb >> 1) + (j - 4));
        ap[qs].w[j] = mylw[widx];
      }

    // O += P @ V ; V fragments (channel-major) shared by both subtiles
    #pragma unroll
    for (int nt = 0; nt < 4; ++nt) {
      const unsigned short* vcol =
          Vb + (size_t)(nt * 16 + r) * NTOK + kt * 32 + kb;
      Frag bv;
      bv.q[0] = *(const uint4*)(vcol);
      bv.q[1] = *(const uint4*)(vcol + 16);
      o[0][nt] = wmma_bf16(ap[0], bv, o[0][nt]);
      o[1][nt] = wmma_bf16(ap[1], bv, o[1][nt]);
    }
  }

  // normalize and write token-major (2304 x 768), channel block h*64
  #pragma unroll
  for (int qs = 0; qs < 2; ++qs) {
    float* orow = out + (size_t)(qt * 32 + qs * 16) * CH + h * HD;
    #pragma unroll
    for (int j = 0; j < 8; ++j) {
      float inv = 1.f / lrow[qs][j];
      int m = j + 8 * hi;
      orow[(size_t)m * CH + r]      = o[qs][0][j] * inv;
      orow[(size_t)m * CH + 16 + r] = o[qs][1][j] * inv;
      orow[(size_t)m * CH + 32 + r] = o[qs][2][j] * inv;
      orow[(size_t)m * CH + 48 + r] = o[qs][3][j] * inv;
    }
  }
}

// ---------------------------------------------------------------------------
extern "C" void kernel_launch(void* const* d_in, const int* in_sizes, int n_in,
                              void* d_out, int out_size, void* d_ws, size_t ws_size,
                              hipStream_t stream) {
  const float* x      = (const float*)d_in[0];
  const float* qkv_w  = (const float*)d_in[1];
  const float* qkv_b  = (const float*)d_in[2];
  const float* FacTu  = (const float*)d_in[3];
  const float* qfacts = (const float*)d_in[4];
  const float* vfacts = (const float*)d_in[5];
  const float* FacTv  = (const float*)d_in[6];
  const float* rph    = (const float*)d_in[7];
  const float* rpw    = (const float*)d_in[8];
  const float* proj_w = (const float*)d_in[9];
  const float* proj_b = (const float*)d_in[10];
  float* outp = (float*)d_out;

  char* ws = (char*)d_ws;
  auto alloc = [&](size_t bytes) {
    char* p = ws;
    ws += (bytes + 255) & ~(size_t)255;
    return p;
  };
  unsigned short* xb   = (unsigned short*)alloc((size_t)NTOK * CH * 2);
  uint32_t*       qwp  = (uint32_t*)alloc((size_t)CH * 3 * CH * 2);       // packed qkv_w
  float*          qkv  = (float*)alloc((size_t)NTOK * 3 * CH * 4);
  float*          u    = (float*)alloc((size_t)NTOK * RANKL * 4);
  float*          uq   = (float*)alloc((size_t)NTOK * RANKL * 4);
  float*          uv   = (float*)alloc((size_t)NTOK * RANKL * 4);
  unsigned short* Qhb  = (unsigned short*)alloc((size_t)NH * NTOK * HD * 2);
  unsigned short* Khb  = (unsigned short*)alloc((size_t)NH * NTOK * HD * 2);
  unsigned short* Vtb  = (unsigned short*)alloc((size_t)NH * HD * NTOK * 2);
  float*          relh = (float*)alloc((size_t)NH * HG * HG * HG * 4);
  float*          relw = (float*)alloc((size_t)NH * HG * HG * HG * 4);
  float*          ao   = (float*)alloc((size_t)NTOK * CH * 4);
  unsigned short* aob  = (unsigned short*)alloc((size_t)NTOK * CH * 2);
  uint32_t*       pwp  = (uint32_t*)alloc((size_t)CH * CH * 2);           // packed proj_w

  // 1) x -> bf16
  k_cvt_bf16<<<(NTOK * CH + 255) / 256, 256, 0, stream>>>(x, xb, NTOK * CH);
  // 2) pack qkv_w (768 x 2304)
  {
    int total = (CH / 32) * ((3 * CH) / 16) * 256;
    k_packB<<<(total + 255) / 256, 256, 0, stream>>>(qkv_w, qwp, CH, 3 * CH);
  }
  // 3) qkv = x @ qkv_w + qkv_b   (2304 x 2304, K=768)
  {
    int waves = (NTOK / 16) * ((3 * CH) / 64);
    k_gemm4<<<waves / 4, 128, 0, stream>>>(xb, qwp, qkv_b, qkv, NTOK, 3 * CH, CH);
  }
  // 4) FacT low-rank path
  k_u<<<(NTOK * RANKL + 255) / 256, 256, 0, stream>>>(x, FacTu, u);
  k_uf<<<(NTOK * RANKL + 255) / 256, 256, 0, stream>>>(u, qfacts, vfacts, uq, uv);
  k_fact_add<<<(NTOK * CH + 255) / 256, 256, 0, stream>>>(uq, uv, FacTv, qkv);
  // 5) split into per-head Q/K (row-major) and V (channel-major), bf16
  k_pack_qkv<<<(NH * NTOK * HD + 255) / 256, 256, 0, stream>>>(qkv, Qhb, Khb, Vtb);
  // 6) decomposed relative position bias tables
  k_rel<<<(NH * HG * HG * HG + 255) / 256, 256, 0, stream>>>(qkv, rph, rpw, relh, relw);
  // 7) fused attention (flash-style, WMMA, 32 q-rows per wave)
  k_attn<<<(NH * NQT2) / 4, 128, 0, stream>>>(Qhb, Khb, Vtb, relh, relw, ao);
  // 8) out = attn_out @ proj_w + proj_b
  k_cvt_bf16<<<(NTOK * CH + 255) / 256, 256, 0, stream>>>(ao, aob, NTOK * CH);
  {
    int total = (CH / 32) * (CH / 16) * 256;
    k_packB<<<(total + 255) / 256, 256, 0, stream>>>(proj_w, pwp, CH, CH);
  }
  {
    int waves = (NTOK / 16) * (CH / 64);
    k_gemm4<<<waves / 4, 128, 0, stream>>>(aob, pwp, proj_b, outp, NTOK, CH, CH);
  }
}
